// LateralMovementGNN_81544249081906
// MI455X (gfx1250) — matrix operand order repaired and checked by reading, not compiled, not run on hardware
//
#include <hip/hip_runtime.h>
#include <hip/hip_fp16.h>

#define N_NODES 100000
#define N_EDGES 1600000
#define N_PRED  1000000
// F_IN=8, HID=64, EMB=32, TDIM=16  -> hcat dim = 80 (pad to 96 for K=32 WMMA steps)

typedef __attribute__((ext_vector_type(16))) _Float16 v16h;
typedef __attribute__((ext_vector_type(8)))  _Float16 v8h;
typedef __attribute__((ext_vector_type(8)))  float    v8f;

// ---------------------------------------------------------------- zero
__global__ void k_zero(float* __restrict__ p, int n) {
    int i = blockIdx.x * blockDim.x + threadIdx.x;
    if (i < n) p[i] = 0.0f;
}

// ---------------------------------------------------------------- layer-1 edge scatter (8 feats + degree)
__global__ void k_scatter1(const float* __restrict__ x, const int* __restrict__ ei,
                           float* __restrict__ agg1, float* __restrict__ deg) {
    int e = blockIdx.x * blockDim.x + threadIdx.x;
    if (e >= N_EDGES) return;
    int s = ei[e];
    int d = ei[N_EDGES + e];
    const float4* xr = (const float4*)(x + (size_t)s * 8);
    float4 v0 = xr[0], v1 = xr[1];
    float* ap = agg1 + (size_t)d * 8;
    unsafeAtomicAdd(ap + 0, v0.x); unsafeAtomicAdd(ap + 1, v0.y);
    unsafeAtomicAdd(ap + 2, v0.z); unsafeAtomicAdd(ap + 3, v0.w);
    unsafeAtomicAdd(ap + 4, v1.x); unsafeAtomicAdd(ap + 5, v1.y);
    unsafeAtomicAdd(ap + 6, v1.z); unsafeAtomicAdd(ap + 7, v1.w);
    unsafeAtomicAdd(deg + d, 1.0f);
}

// ---------------------------------------------------------------- layer-1 node update: h = relu(agg/deg @ W1n.T + b1 + x @ W1r.T)
__global__ __launch_bounds__(64) void k_layer1(const float* __restrict__ x, const float* __restrict__ agg1,
                         const float* __restrict__ deg, const float* __restrict__ W1n,
                         const float* __restrict__ b1, const float* __restrict__ W1r,
                         float* __restrict__ h) {
    __shared__ float xs[8], as[8];
    int n = blockIdx.x;
    int j = threadIdx.x;              // 0..63
    if (j < 8) { xs[j] = x[(size_t)n * 8 + j]; as[j] = agg1[(size_t)n * 8 + j]; }
    __syncthreads();
    float dinv = 1.0f / fmaxf(deg[n], 1.0f);
    float acc = b1[j];
#pragma unroll
    for (int k = 0; k < 8; ++k)
        acc += as[k] * dinv * W1n[j * 8 + k] + xs[k] * W1r[j * 8 + k];
    h[(size_t)n * 64 + j] = fmaxf(acc, 0.0f);
}

// ---------------------------------------------------------------- layer-2 edge scatter (64 feats)
__global__ void k_scatter2(const float* __restrict__ h, const int* __restrict__ ei,
                           float* __restrict__ agg2) {
    int idx = blockIdx.x * blockDim.x + threadIdx.x;
    if (idx >= N_EDGES * 8) return;
    int e = idx >> 3;
    int c = idx & 7;
    int s = ei[e];
    int d = ei[N_EDGES + e];
    const float4* hp = (const float4*)(h + (size_t)s * 64 + c * 8);
    float4 v0 = hp[0], v1 = hp[1];
    float* ap = agg2 + (size_t)d * 64 + c * 8;
    unsafeAtomicAdd(ap + 0, v0.x); unsafeAtomicAdd(ap + 1, v0.y);
    unsafeAtomicAdd(ap + 2, v0.z); unsafeAtomicAdd(ap + 3, v0.w);
    unsafeAtomicAdd(ap + 4, v1.x); unsafeAtomicAdd(ap + 5, v1.y);
    unsafeAtomicAdd(ap + 6, v1.z); unsafeAtomicAdd(ap + 7, v1.w);
}

// ---------------------------------------------------------------- layer-2 node update: z = agg/deg @ W2n.T + b2 + h @ W2r.T  (store f16)
__global__ __launch_bounds__(64) void k_layer2(const float* __restrict__ h, const float* __restrict__ agg2,
                         const float* __restrict__ deg, const float* __restrict__ W2n,
                         const float* __restrict__ b2, const float* __restrict__ W2r,
                         _Float16* __restrict__ z16) {
    __shared__ float hs[64], as[64];
    int n = blockIdx.x;
    int t = threadIdx.x;              // 0..63
    hs[t] = h[(size_t)n * 64 + t];
    as[t] = agg2[(size_t)n * 64 + t];
    __syncthreads();
    if (t < 32) {
        float dinv = 1.0f / fmaxf(deg[n], 1.0f);
        float acc = b2[t];
#pragma unroll 8
        for (int k = 0; k < 64; ++k)
            acc += as[k] * dinv * W2n[t * 64 + k] + hs[k] * W2r[t * 64 + k];
        z16[(size_t)n * 32 + t] = (_Float16)acc;
    }
}

// ---------------------------------------------------------------- prediction MLP (all four matmuls on WMMA, f32 accum)
// Per wave: tiles of 16 pred edges.
//   time MLP : A built directly in A-fragment registers, 1 WMMA (tb2 via C)
//   layer 1  : 16x96 @ 96x64  -> 12 WMMA
//   layer 2  : 16x64 @ 64x32  ->  4 WMMA
//   layer 3  : 16x32 @ 32x[1 in col 0] -> 1 WMMA, column 0 of D stored
__global__ __launch_bounds__(256) void k_pred(
    const _Float16* __restrict__ z16, const int* __restrict__ pe, const float* __restrict__ ts,
    const float* __restrict__ tW1, const float* __restrict__ tb1,
    const float* __restrict__ tW2, const float* __restrict__ tb2,
    const float* __restrict__ pW1, const float* __restrict__ pb1,
    const float* __restrict__ pW2, const float* __restrict__ pb2,
    const float* __restrict__ pW3, const float* __restrict__ pb3,
    float* __restrict__ out) {
    __shared__ _Float16 sW1t[96][64];          // B layer1: [k][n] = pW1[n][k]
    __shared__ _Float16 sW2t[64][32];          // B layer2: [k][n] = pW2[n][k]
    __shared__ _Float16 sTW2t[32][16];         // B time:   [k][n] = tW2[n][k]
    __shared__ float sTW1[32], sTB1[32], sTB2[16], sPB1[64], sPB2[32];
    __shared__ _Float16 sHcat[8][16][96];      // per-wave layer-1 input tile
    __shared__ _Float16 sAct1[8][16][64];      // per-wave layer-1 activations
    __shared__ _Float16 sAct2[8][16][32];      // per-wave layer-2 activations

    const int tid = threadIdx.x;
    // ---- stage weights ----
    for (int idx = tid; idx < 96 * 64; idx += 256) {
        int k = idx >> 6, n = idx & 63;
        sW1t[k][n] = (k < 80) ? (_Float16)pW1[n * 80 + k] : (_Float16)0.0f;
    }
    for (int idx = tid; idx < 64 * 32; idx += 256) {
        int k = idx >> 5, n = idx & 31;
        sW2t[k][n] = (_Float16)pW2[n * 64 + k];
    }
    if (tid < 512) {
        int k = tid >> 4, n = tid & 15;
        sTW2t[k][n] = (_Float16)tW2[n * 32 + k];
    }
    if (tid < 32) { sTW1[tid] = tW1[tid]; sTB1[tid] = tb1[tid]; sPB2[tid] = pb2[tid]; }
    if (tid < 64) sPB1[tid] = pb1[tid];
    if (tid < 16) sTB2[tid] = tb2[tid];
    const float bias3 = pb3[0];
    __syncthreads();

    const int lane = tid & 31;
    const int wv   = tid >> 5;
    const int n15  = lane & 15;          // N index (B/C/D) / M row (A)
    const int hi   = lane >> 4;          // lane half
    const int khiA = hi * 8;             // A-fragment K sub-offset
    const int kbB  = hi * 16;            // unused; B fragments take K from khiA
    const int mrow = hi * 8;             // C/D M base

    // ---- load B fragments into registers (B 32x16: N = lane%16, K = (elem<8?0:16) + lane/16*8 + elem%8) ----
    v16h bf1[3][4];
#pragma unroll
    for (int kc = 0; kc < 3; ++kc)
#pragma unroll
        for (int nc = 0; nc < 4; ++nc)
#pragma unroll
            for (int i = 0; i < 16; ++i)
                bf1[kc][nc][i] = sW1t[kc * 32 + khiA + ((i & 8) << 1) + (i & 7)][nc * 16 + n15];
    v16h bf2[2][2];
#pragma unroll
    for (int kc = 0; kc < 2; ++kc)
#pragma unroll
        for (int nc = 0; nc < 2; ++nc)
#pragma unroll
            for (int i = 0; i < 16; ++i)
                bf2[kc][nc][i] = sW2t[kc * 32 + khiA + ((i & 8) << 1) + (i & 7)][nc * 16 + n15];
    v16h bft;                            // time MLP B
#pragma unroll
    for (int i = 0; i < 16; ++i) bft[i] = sTW2t[khiA + ((i & 8) << 1) + (i & 7)][n15];
    v16h bf3;                            // layer3 B: pW3 in column N=0, zeros elsewhere
#pragma unroll
    for (int i = 0; i < 16; ++i)
        bf3[i] = (n15 == 0) ? (_Float16)pW3[khiA + ((i & 8) << 1) + (i & 7)] : (_Float16)0.0f;

    _Float16 (*hc)[96] = sHcat[wv];
    _Float16 (*a1)[64] = sAct1[wv];
    _Float16 (*a2)[32] = sAct2[wv];
    const int m = n15;

    // ---- zero the K-pad columns (80..95) once; never overwritten ----
    if (hi == 0) {
        v8h zz = {};
        *(v8h*)&hc[m][80] = zz;
        *(v8h*)&hc[m][88] = zz;
    }

    const int ntiles = N_PRED / 16;
    const int stride = gridDim.x * 8;
    for (int tile = blockIdx.x * 8 + wv; tile < ntiles; tile += stride) {
        const int e0 = tile * 16;
        // ---- gather z_src (lanes 0-15) / z_dst (lanes 16-31): 32 halves each ----
        {
            int edge = e0 + m;
            int node = (hi == 0) ? pe[edge] : pe[N_PRED + edge];
            const unsigned int* zr = (const unsigned int*)(z16 + (size_t)node * 32);
            unsigned int* dp = (unsigned int*)&hc[m][hi ? 32 : 0];
#pragma unroll
            for (int w = 0; w < 16; ++w) dp[w] = zr[w];
        }
        // ---- time MLP: hidden relu(t*tW1+tb1) built straight into A-fragment regs ----
        {
            float t = ts[e0 + m];        // all lanes: row m = lane&15
            v16h ta;
#pragma unroll
            for (int i = 0; i < 16; ++i) {
                int k = ((i & 8) << 1) + khiA + (i & 7);   // (i<8?0:16)+khiA+(i%8)
                float v = t * sTW1[k] + sTB1[k];
                ta[i] = (_Float16)(v > 0.0f ? v : 0.0f);
            }
            v8f tc;
#pragma unroll
            for (int v = 0; v < 8; ++v) tc[v] = sTB2[n15];      // + tb2 via C
            tc = __builtin_amdgcn_wmma_f32_16x16x32_f16(false, ta, false, bft, (short)0, tc, false, false);
#pragma unroll
            for (int v = 0; v < 8; ++v)
                hc[v + mrow][64 + n15] = (_Float16)tc[v];       // D: M=v+mrow, N=n15
        }
        // per-wave LDS write->read: DS ops from one wave are in-order.

        // ---- layer 1: 16x96 @ 96x64 via 3 K-steps x 4 N-tiles ----
        v8f c0 = {}, c1 = {}, c2 = {}, c3 = {};
#pragma unroll
        for (int kc = 0; kc < 3; ++kc) {
            const int k0 = kc * 32;
            v8h lo = *(const v8h*)&hc[m][k0 + khiA];
            v8h hh = *(const v8h*)&hc[m][k0 + 16 + khiA];
            v16h a = __builtin_shufflevector(lo, hh, 0,1,2,3,4,5,6,7,8,9,10,11,12,13,14,15);
            c0 = __builtin_amdgcn_wmma_f32_16x16x32_f16(false, a, false, bf1[kc][0], (short)0, c0, false, false);
            c1 = __builtin_amdgcn_wmma_f32_16x16x32_f16(false, a, false, bf1[kc][1], (short)0, c1, false, false);
            c2 = __builtin_amdgcn_wmma_f32_16x16x32_f16(false, a, false, bf1[kc][2], (short)0, c2, false, false);
            c3 = __builtin_amdgcn_wmma_f32_16x16x32_f16(false, a, false, bf1[kc][3], (short)0, c3, false, false);
        }
#pragma unroll
        for (int v = 0; v < 8; ++v) {
            int M = v + mrow;
            float t0 = c0[v] + sPB1[n15];
            float t1 = c1[v] + sPB1[16 + n15];
            float t2 = c2[v] + sPB1[32 + n15];
            float t3 = c3[v] + sPB1[48 + n15];
            a1[M][n15]      = (_Float16)(t0 > 0.0f ? t0 : 0.0f);
            a1[M][16 + n15] = (_Float16)(t1 > 0.0f ? t1 : 0.0f);
            a1[M][32 + n15] = (_Float16)(t2 > 0.0f ? t2 : 0.0f);
            a1[M][48 + n15] = (_Float16)(t3 > 0.0f ? t3 : 0.0f);
        }

        // ---- layer 2: 16x64 @ 64x32 via 2 K-steps x 2 N-tiles ----
        v8f d0 = {}, d1 = {};
#pragma unroll
        for (int kc = 0; kc < 2; ++kc) {
            const int k0 = kc * 32;
            v8h lo = *(const v8h*)&a1[m][k0 + khiA];
            v8h hh = *(const v8h*)&a1[m][k0 + 16 + khiA];
            v16h a = __builtin_shufflevector(lo, hh, 0,1,2,3,4,5,6,7,8,9,10,11,12,13,14,15);
            d0 = __builtin_amdgcn_wmma_f32_16x16x32_f16(false, a, false, bf2[kc][0], (short)0, d0, false, false);
            d1 = __builtin_amdgcn_wmma_f32_16x16x32_f16(false, a, false, bf2[kc][1], (short)0, d1, false, false);
        }
#pragma unroll
        for (int v = 0; v < 8; ++v) {
            int M = v + mrow;
            float t0 = d0[v] + sPB2[n15];
            float t1 = d1[v] + sPB2[16 + n15];
            a2[M][n15]      = (_Float16)(t0 > 0.0f ? t0 : 0.0f);
            a2[M][16 + n15] = (_Float16)(t1 > 0.0f ? t1 : 0.0f);
        }

        // ---- layer 3: 16x32 @ 32x16 (pW3 in column 0) ----
        {
            v8h lo = *(const v8h*)&a2[m][khiA];
            v8h hh = *(const v8h*)&a2[m][16 + khiA];
            v16h a = __builtin_shufflevector(lo, hh, 0,1,2,3,4,5,6,7,8,9,10,11,12,13,14,15);
            v8f f = {};
            f = __builtin_amdgcn_wmma_f32_16x16x32_f16(false, a, false, bf3, (short)0, f, false, false);
            if (n15 == 0) {              // lanes 0 & 16 hold column 0 -> rows mrow..mrow+7
                float4 r0 = make_float4(f[0] + bias3, f[1] + bias3, f[2] + bias3, f[3] + bias3);
                float4 r1 = make_float4(f[4] + bias3, f[5] + bias3, f[6] + bias3, f[7] + bias3);
                *(float4*)(out + e0 + mrow)     = r0;
                *(float4*)(out + e0 + mrow + 4) = r1;
            }
        }
    }
}

// ---------------------------------------------------------------- host
extern "C" void kernel_launch(void* const* d_in, const int* in_sizes, int n_in,
                              void* d_out, int out_size, void* d_ws, size_t ws_size,
                              hipStream_t stream) {
    const float* x   = (const float*)d_in[0];
    const int*   ei  = (const int*)d_in[1];
    const int*   pe  = (const int*)d_in[2];
    const float* ts  = (const float*)d_in[3];
    const float* W1n = (const float*)d_in[4];
    const float* b1  = (const float*)d_in[5];
    const float* W1r = (const float*)d_in[6];
    const float* W2n = (const float*)d_in[7];
    const float* b2  = (const float*)d_in[8];
    const float* W2r = (const float*)d_in[9];
    const float* tW1 = (const float*)d_in[10];
    const float* tb1 = (const float*)d_in[11];
    const float* tW2 = (const float*)d_in[12];
    const float* tb2 = (const float*)d_in[13];
    const float* pW1 = (const float*)d_in[14];
    const float* pb1 = (const float*)d_in[15];
    const float* pW2 = (const float*)d_in[16];
    const float* pb2 = (const float*)d_in[17];
    const float* pW3 = (const float*)d_in[18];
    const float* pb3 = (const float*)d_in[19];
    float* out = (float*)d_out;

    // workspace carve-up (floats): agg1[800000] deg[100000] agg2[6400000] h[6400000] z16[3.2M halves]
    float* agg1 = (float*)d_ws;
    float* deg  = agg1 + 800000;
    float* agg2 = deg + 100000;
    float* h    = agg2 + 6400000;
    _Float16* z16 = (_Float16*)(h + 6400000);

    const int zeroN = 800000 + 100000 + 6400000;   // agg1+deg+agg2 contiguous
    k_zero<<<(zeroN + 255) / 256, 256, 0, stream>>>(agg1, zeroN);
    k_scatter1<<<(N_EDGES + 255) / 256, 256, 0, stream>>>(x, ei, agg1, deg);
    k_layer1<<<N_NODES, 64, 0, stream>>>(x, agg1, deg, W1n, b1, W1r, h);
    k_scatter2<<<(N_EDGES * 8 + 255) / 256, 256, 0, stream>>>(h, ei, agg2);
    k_layer2<<<N_NODES, 64, 0, stream>>>(h, agg2, deg, W2n, b2, W2r, z16);
    k_pred<<<512, 256, 0, stream>>>(z16, pe, ts, tW1, tb1, tW2, tb2,
                                    pW1, pb1, pW2, pb2, pW3, pb3, out);
}
